// QuantLinear_24721831756582
// MI455X (gfx1250) — compile-verified, matrix operand order checked by reading, not run on hardware
//
#include <hip/hip_runtime.h>
#include <stdint.h>

#define IN_F   4096
#define OUT_F  4096
#define GROUP  128
#define BM     128
#define BN     128
#define BK     32
#define LDK    40      // padded LDS K-stride in halves: 80B = 20 dwords, conflict-free
#define THREADS 256

typedef __attribute__((ext_vector_type(16))) _Float16 v16h;
typedef __attribute__((ext_vector_type(8)))  _Float16 v8h;
typedef __attribute__((ext_vector_type(2)))  _Float16 h2;
typedef __attribute__((ext_vector_type(8)))  float    v8f;
typedef __attribute__((ext_vector_type(4)))  float    v4f;
typedef __attribute__((ext_vector_type(4)))  int      v4i;

__global__ __launch_bounds__(THREADS)
void qlinear_wmma_f16(const float* __restrict__ x,
                      const int*   __restrict__ packed,
                      const float* __restrict__ scales,
                      const float* __restrict__ offsets,
                      const float* __restrict__ bias,
                      float*       __restrict__ out)
{
    __shared__ _Float16 As[2][BM][LDK];
    __shared__ _Float16 Bs[2][BN][LDK];

    const int tid  = threadIdx.x;
    const int lane = tid & 31;              // wave32
    const int wave = tid >> 5;              // 0..7
    const int wm   = (wave >> 2) * 64;      // wave row base within block tile
    const int wn   = (wave & 3)  * 32;      // wave col base within block tile

    const int m0 = blockIdx.y * BM;
    const int n0 = blockIdx.x * BN;

    // loader mapping: 2 threads per tile row, each handles 16 K-elements
    const int lrow  = tid >> 1;             // 0..127
    const int lhalf = tid & 1;              // which 16-wide half of the 32-wide K slab

    // fragment lane mapping per ISA 16-bit A/B layout
    const int fr = lane & 15;               // row (A) / col (B) within 16x16 tile
    const int kb = (lane >> 4) << 3;        // K phase: 0 or 8

    v8f acc[4][2];
    #pragma unroll
    for (int i = 0; i < 4; ++i) {
        #pragma unroll
        for (int j = 0; j < 2; ++j) { v8f z = {}; acc[i][j] = z; }
    }

    // staging registers (global -> regs -> LDS)
    v4f   ax[4];
    v4i   bp[2];
    float qscale, qoff;

    auto load_global = [&](int kt) {
        const int k0 = kt * BK;
        // A tile: x[m0+lrow][k0 + lhalf*16 .. +15]  (fp32)
        const float* ap = x + (size_t)(m0 + lrow) * IN_F + k0 + lhalf * 16;
        ax[0] = *(const v4f*)(ap + 0);
        ax[1] = *(const v4f*)(ap + 4);
        ax[2] = *(const v4f*)(ap + 8);
        ax[3] = *(const v4f*)(ap + 12);
        // B tile: W[n0+lrow][k0 + lhalf*16 .. +15]  (int4: one byte per int32 elem)
        const int    ng   = n0 + lrow;
        const size_t flat = (size_t)ng * IN_F + (size_t)(k0 + lhalf * 16);
        const int*   pp   = packed + (flat >> 1);
        bp[0] = *(const v4i*)(pp + 0);
        bp[1] = *(const v4i*)(pp + 4);
        // 32-wide K slab lies inside one 128-wide group (k0 % 128 in {0,32,64,96})
        const int g = ng * (IN_F / GROUP) + (k0 >> 7);
        qscale = scales[g];
        qoff   = offsets[g];
    };

    auto store_lds = [&](int buf) {
        // ---- A: fp32 -> f16 (packed converts) ----
        v8h h0, h1;
        #pragma unroll
        for (int e = 0; e < 8; ++e) {
            h0[e] = (_Float16)ax[(e >> 2)    ][e & 3];
            h1[e] = (_Float16)ax[(e >> 2) + 2][e & 3];
        }
        _Float16* da = &As[buf][lrow][lhalf * 16];
        *(v8h*)(da + 0) = h0;
        *(v8h*)(da + 8) = h1;

        // ---- B: int4 dequant via f16 magic-number, exact variant ----
        // 0x6400|q == f16(1024+q) exactly; (1024+q) - 1024 == q exactly in f16;
        // then one packed FMA: w = q*s + o (single rounding).
        h2 sc, of, kc;
        sc[0] = sc[1] = (_Float16)qscale;
        of[0] = of[1] = (_Float16)qoff;
        kc[0] = kc[1] = (_Float16)(-1024.0f);

        v8h w0, w1;
        #pragma unroll
        for (int i = 0; i < 4; ++i) {
            const uint32_t p0 = (uint32_t)bp[0][i];
            const uint32_t p1 = (uint32_t)bp[1][i];
            const uint32_t m0b = 0x64006400u | (p0 & 0xFu) | ((p0 & 0xF0u) << 12);
            const uint32_t m1b = 0x64006400u | (p1 & 0xFu) | ((p1 & 0xF0u) << 12);
            h2 q0 = __builtin_bit_cast(h2, m0b) + kc;   // v_pk_add_f16 (exact)
            h2 q1 = __builtin_bit_cast(h2, m1b) + kc;
            h2 wv0 = q0 * sc + of;                      // v_pk_fma_f16
            h2 wv1 = q1 * sc + of;
            w0[2*i + 0] = wv0[0];
            w0[2*i + 1] = wv0[1];
            w1[2*i + 0] = wv1[0];
            w1[2*i + 1] = wv1[1];
        }
        _Float16* db = &Bs[buf][lrow][lhalf * 16];
        *(v8h*)(db + 0) = w0;
        *(v8h*)(db + 8) = w1;
    };

    // load one 16x32 fragment (A) or 32x16^T fragment (B) from row-major [16][LDK] LDS
    auto load_frag = [&](const _Float16* rowptr) -> v16h {
        v8h lo = *(const v8h*)(rowptr + kb);         // K = kb..kb+7
        v8h hi = *(const v8h*)(rowptr + 16 + kb);    // K = 16+kb..16+kb+7
        return __builtin_shufflevector(lo, hi,
                0, 1, 2, 3, 4, 5, 6, 7, 8, 9, 10, 11, 12, 13, 14, 15);
    };

    auto compute = [&](int buf) {
        // fetch ALL fragments first so the scheduler can batch ds waits,
        // then issue 8 back-to-back WMMAs
        v16h bfrag[2], afrag[4];
        #pragma unroll
        for (int tn = 0; tn < 2; ++tn)
            bfrag[tn] = load_frag(&Bs[buf][wn + tn * 16 + fr][0]);
        #pragma unroll
        for (int tm = 0; tm < 4; ++tm)
            afrag[tm] = load_frag(&As[buf][wm + tm * 16 + fr][0]);
        #pragma unroll
        for (int tm = 0; tm < 4; ++tm)
            #pragma unroll
            for (int tn = 0; tn < 2; ++tn)
                acc[tm][tn] = __builtin_amdgcn_wmma_f32_16x16x32_f16(
                    false, afrag[tm], false, bfrag[tn],
                    (short)0, acc[tm][tn], false, false);
    };

    const int nk = IN_F / BK;   // 128 K-steps

    load_global(0);
    store_lds(0);
    __syncthreads();

    for (int kt = 0; kt < nk; ++kt) {
        const int buf = kt & 1;
        if (kt + 1 < nk) load_global(kt + 1);
        if (kt + 4 < nk) {
            const int kp = (kt + 4) * BK;
            __builtin_prefetch(x + (size_t)(m0 + lrow) * IN_F + kp + lhalf * 16, 0, 1);
            __builtin_prefetch(packed + (((size_t)(n0 + lrow) * IN_F + kp + lhalf * 16) >> 1), 0, 1);
        }
        compute(buf);
        if (kt + 1 < nk) store_lds(buf ^ 1);
        __syncthreads();
    }

    // Epilogue: C layout = VGPR r -> M = r + 8*(lane>=16), N = lane&15
    const int mr = (lane >> 4) << 3;
    #pragma unroll
    for (int tm = 0; tm < 4; ++tm) {
        #pragma unroll
        for (int tn = 0; tn < 2; ++tn) {
            const int   n  = n0 + wn + tn * 16 + fr;
            const float bv = bias[n];
            const int   mb = m0 + wm + tm * 16 + mr;
            float* op = out + (size_t)mb * OUT_F + n;
            #pragma unroll
            for (int r = 0; r < 8; ++r)
                op[(size_t)r * OUT_F] = acc[tm][tn][r] + bv;
        }
    }
}

extern "C" void kernel_launch(void* const* d_in, const int* in_sizes, int n_in,
                              void* d_out, int out_size, void* d_ws, size_t ws_size,
                              hipStream_t stream) {
    const float* x       = (const float*)d_in[0];
    const int*   packed  = (const int*)  d_in[1];
    const float* scales  = (const float*)d_in[2];
    const float* offsets = (const float*)d_in[3];
    const float* bias    = (const float*)d_in[4];
    float*       out     = (float*)d_out;

    const int M = in_sizes[0] / IN_F;          // 4*2048 = 8192
    dim3 grid(OUT_F / BN, M / BM);             // 32 x 64 = 2048 blocks
    qlinear_wmma_f16<<<grid, THREADS, 0, stream>>>(x, packed, scales, offsets, bias, out);
}
